// AttentionBlock_48000554500804
// MI455X (gfx1250) — matrix operand order repaired and checked by reading, not compiled, not run on hardware
//
#include <hip/hip_runtime.h>
#include <hip/hip_bf16.h>

typedef __attribute__((ext_vector_type(16))) __bf16 v16bf;
typedef __attribute__((ext_vector_type(8)))  float  v8f;

#define WPB 4           // waves per block (wave32)
#define TPB (WPB * 32)

// C[M x 64] = A[M x 768] @ W[64 x 768]^T  (bf16 WMMA, fp32 accum), 2 M-tiles/wave.
//   mode 0: out[M x 64] = l2norm(row + bias)
//   mode 1: out[M]      = xfeat[row/64] . l2norm(row + bias)   (cosine scores)
__global__ __launch_bounds__(TPB)
void feat_gemm_kernel(const float* __restrict__ A,
                      const float* __restrict__ W,
                      const float* __restrict__ bias,
                      const float* __restrict__ xfeat,
                      float* __restrict__ out,
                      int mode)
{
    // staged half of W as bf16: 64 rows x 384 cols, padded row stride 392 (bank-conflict-free)
    __shared__ __bf16 sW[64 * 392];   // 50176 bytes

    const int lane  = threadIdx.x & 31;
    const int wave  = threadIdx.x >> 5;
    const int tile2 = blockIdx.x * WPB + wave;   // 32-row M super-tile
    const int r     = lane & 15;
    const int half  = lane >> 4;

    v8f acc[2][4];
#pragma unroll
    for (int t = 0; t < 2; ++t)
#pragma unroll
        for (int nt = 0; nt < 4; ++nt)
            acc[t][nt] = (v8f){0.f, 0.f, 0.f, 0.f, 0.f, 0.f, 0.f, 0.f};

    // A-operand rows (16-bit A layout: lane r = row r, K = half*8 + {0..7,16..23})
    const float* arow0 = A + (size_t)(tile2 * 32 + r) * 768 + half * 8;
    const float* arow1 = arow0 + (size_t)16 * 768;

    for (int stage = 0; stage < 2; ++stage) {
        __syncthreads();
        // cooperative load + fp32->bf16 convert of W[:, stage*384 : stage*384+384]
        for (int it = 0; it < 6144 / TPB; ++it) {
            int fidx = it * TPB + threadIdx.x;     // float4 index, 96 per row
            int row  = fidx / 96;
            int c4   = (fidx % 96) * 4;
            float4 f = *(const float4*)(W + (size_t)row * 768 + stage * 384 + c4);
            __bf16* dst = &sW[row * 392 + c4];
            dst[0] = (__bf16)f.x; dst[1] = (__bf16)f.y;
            dst[2] = (__bf16)f.z; dst[3] = (__bf16)f.w;
        }
        __syncthreads();

#pragma unroll
        for (int kc = 0; kc < 12; ++kc) {
            // ---- two A tiles (16x32 fp32 -> bf16, WMMA A layout) ----
            const float* rp0 = arow0 + stage * 384 + kc * 32;
            const float* rp1 = arow1 + stage * 384 + kc * 32;
            float4 f0 = ((const float4*)rp0)[0];
            float4 f1 = ((const float4*)rp0)[1];
            float4 f2 = ((const float4*)(rp0 + 16))[0];
            float4 f3 = ((const float4*)(rp0 + 16))[1];
            float4 g0 = ((const float4*)rp1)[0];
            float4 g1 = ((const float4*)rp1)[1];
            float4 g2 = ((const float4*)(rp1 + 16))[0];
            float4 g3 = ((const float4*)(rp1 + 16))[1];

            union { v16bf v; __bf16 h[16]; } av0, av1;
            av0.h[0]  = (__bf16)f0.x; av0.h[1]  = (__bf16)f0.y;
            av0.h[2]  = (__bf16)f0.z; av0.h[3]  = (__bf16)f0.w;
            av0.h[4]  = (__bf16)f1.x; av0.h[5]  = (__bf16)f1.y;
            av0.h[6]  = (__bf16)f1.z; av0.h[7]  = (__bf16)f1.w;
            av0.h[8]  = (__bf16)f2.x; av0.h[9]  = (__bf16)f2.y;
            av0.h[10] = (__bf16)f2.z; av0.h[11] = (__bf16)f2.w;
            av0.h[12] = (__bf16)f3.x; av0.h[13] = (__bf16)f3.y;
            av0.h[14] = (__bf16)f3.z; av0.h[15] = (__bf16)f3.w;
            av1.h[0]  = (__bf16)g0.x; av1.h[1]  = (__bf16)g0.y;
            av1.h[2]  = (__bf16)g0.z; av1.h[3]  = (__bf16)g0.w;
            av1.h[4]  = (__bf16)g1.x; av1.h[5]  = (__bf16)g1.y;
            av1.h[6]  = (__bf16)g1.z; av1.h[7]  = (__bf16)g1.w;
            av1.h[8]  = (__bf16)g2.x; av1.h[9]  = (__bf16)g2.y;
            av1.h[10] = (__bf16)g2.z; av1.h[11] = (__bf16)g2.w;
            av1.h[12] = (__bf16)g3.x; av1.h[13] = (__bf16)g3.y;
            av1.h[14] = (__bf16)g3.z; av1.h[15] = (__bf16)g3.w;

            // ---- preload all four B tiles (distinct regs -> batched ds_loads) ----
            union { v16bf v; uint4 u[2]; } bv[4];
#pragma unroll
            for (int nt = 0; nt < 4; ++nt) {
                const uint4* bp = (const uint4*)&sW[(nt * 16 + r) * 392 + kc * 32 + half * 8];
                bv[nt].u[0] = bp[0];
                bv[nt].u[1] = bp[2];
            }
#pragma unroll
            for (int nt = 0; nt < 4; ++nt) {
                acc[0][nt] = __builtin_amdgcn_wmma_f32_16x16x32_bf16(
                    false, av0.v, false, bv[nt].v, (short)0, acc[0][nt], false, false);
                acc[1][nt] = __builtin_amdgcn_wmma_f32_16x16x32_bf16(
                    false, av1.v, false, bv[nt].v, (short)0, acc[1][nt], false, false);
            }
        }
    }

    // ---- epilogue ----
    // C/D layout: lane owns column col(nt) = nt*16 + r; VGPR j holds row j + 8*half
    float bcol[4], xcol[4];
#pragma unroll
    for (int nt = 0; nt < 4; ++nt) {
        int col = nt * 16 + r;
        bcol[nt] = bias[col];
        xcol[nt] = (mode == 1) ? xfeat[(size_t)(tile2 >> 1) * 64 + col] : 0.f;
    }
#pragma unroll
    for (int t = 0; t < 2; ++t) {
#pragma unroll
        for (int j = 0; j < 8; ++j) {
            float v0 = acc[t][0][j] + bcol[0];
            float v1 = acc[t][1][j] + bcol[1];
            float v2 = acc[t][2][j] + bcol[2];
            float v3 = acc[t][3][j] + bcol[3];
            float sq = v0 * v0 + v1 * v1 + v2 * v2 + v3 * v3;
            float xp = (mode == 1) ? (v0 * xcol[0] + v1 * xcol[1] + v2 * xcol[2] + v3 * xcol[3]) : 0.f;
#pragma unroll
            for (int off = 1; off < 16; off <<= 1) {
                sq += __shfl_xor(sq, off, 32);
                xp += __shfl_xor(xp, off, 32);
            }
            int row = tile2 * 32 + t * 16 + j + 8 * half;
            float inv = 1.0f / fmaxf(sqrtf(sq), 1e-12f);
            if (mode == 0) {
                float* o = out + (size_t)row * 64;
                o[r]      = v0 * inv;
                o[16 + r] = v1 * inv;
                o[32 + r] = v2 * inv;
                o[48 + r] = v3 * inv;
            } else {
                if (r == j) out[row] = xp * inv;
            }
        }
    }
}

// one wave per batch row: max -> switch, softmax(sharpness * scores)
__global__ __launch_bounds__(256)
void softmax_kernel(const float* __restrict__ scores,
                    const float* __restrict__ sig_scale,
                    const float* __restrict__ sig_shift,
                    float* __restrict__ weights,
                    float* __restrict__ swv)
{
    const int lane = threadIdx.x & 31;
    const int b    = blockIdx.x * 8 + (threadIdx.x >> 5);
    const float* s = scores + (size_t)b * 64;
    float s0 = s[lane];
    float s1 = s[lane + 32];
    float m = fmaxf(s0, s1);
#pragma unroll
    for (int off = 1; off < 32; off <<= 1) m = fmaxf(m, __shfl_xor(m, off, 32));
    const float sharp = 1048576.0f;  // CF_FEAT * E^3 * 4
    float e0 = expf(sharp * (s0 - m));
    float e1 = expf(sharp * (s1 - m));
    float sum = e0 + e1;
#pragma unroll
    for (int off = 1; off < 32; off <<= 1) sum += __shfl_xor(sum, off, 32);
    float inv = 1.0f / sum;
    weights[(size_t)b * 64 + lane]      = e0 * inv;
    weights[(size_t)b * 64 + lane + 32] = e1 * inv;
    if (lane == 0) {
        float z = m * sig_scale[0] + sig_shift[0];
        swv[b] = 1.0f / (1.0f + expf(-z));
    }
}

// per-b block, 256 threads = 256 pixels; skips exact-zero weights (bit-identical)
__global__ __launch_bounds__(256)
void output_kernel(const float* __restrict__ x,  const float* __restrict__ p,
                   const float* __restrict__ weights, const float* __restrict__ swv,
                   const float* __restrict__ Wg, const float* __restrict__ bg,
                   const float* __restrict__ Wo, const float* __restrict__ bo,
                   float* __restrict__ out)
{
    __shared__ float wts[64];
    const int b   = blockIdx.x;
    const int pix = threadIdx.x;
    if (threadIdx.x < 64) wts[threadIdx.x] = weights[(size_t)b * 64 + threadIdx.x];
    __syncthreads();

    const float g00 = Wg[0], g01 = Wg[1], g02 = Wg[2];
    const float g10 = Wg[3], g11 = Wg[4], g12 = Wg[5];
    const float g20 = Wg[6], g21 = Wg[7], g22 = Wg[8];
    const float bg0 = bg[0], bg1 = bg[1], bg2 = bg[2];

    float a0 = 0.f, a1 = 0.f, a2 = 0.f;
    const float* pb = p + (size_t)b * 64 * 768;
    for (int k = 0; k < 64; ++k) {
        float wk = wts[k];
        if (wk != 0.0f) {   // uniform branch; zero terms contribute exactly 0
            const float* pk = pb + (size_t)k * 768;
            float p0 = pk[pix], p1 = pk[256 + pix], p2 = pk[512 + pix];
            a0 += wk * (g00 * p0 + g01 * p1 + g02 * p2 + bg0);
            a1 += wk * (g10 * p0 + g11 * p1 + g12 * p2 + bg1);
            a2 += wk * (g20 * p0 + g21 * p1 + g22 * p2 + bg2);
        }
    }
    const float sw  = swv[b];
    const float omw = 1.0f - sw;
    float pa0 = Wo[0] * a0 + Wo[1] * a1 + Wo[2] * a2 + bo[0];
    float pa1 = Wo[3] * a0 + Wo[4] * a1 + Wo[5] * a2 + bo[1];
    float pa2 = Wo[6] * a0 + Wo[7] * a1 + Wo[8] * a2 + bo[2];
    const float* xb = x + (size_t)b * 768;
    float* ob = out + (size_t)b * 768;
    ob[pix]       = xb[pix]       * omw + pa0 * sw;
    ob[256 + pix] = xb[256 + pix] * omw + pa1 * sw;
    ob[512 + pix] = xb[512 + pix] * omw + pa2 * sw;
}

extern "C" void kernel_launch(void* const* d_in, const int* in_sizes, int n_in,
                              void* d_out, int out_size, void* d_ws, size_t ws_size,
                              hipStream_t stream) {
    const float* x      = (const float*)d_in[0];
    const float* p      = (const float*)d_in[1];
    const float* x_im   = (const float*)d_in[2];
    const float* p_im   = (const float*)d_in[3];
    const float* Wtheta = (const float*)d_in[4];
    const float* btheta = (const float*)d_in[5];
    const float* Wphi   = (const float*)d_in[6];
    const float* bphi   = (const float*)d_in[7];
    const float* Wg     = (const float*)d_in[8];
    const float* bg     = (const float*)d_in[9];
    const float* Wo     = (const float*)d_in[10];
    const float* bo     = (const float*)d_in[11];
    const float* sscale = (const float*)d_in[12];
    const float* sshift = (const float*)d_in[13];
    float* out = (float*)d_out;

    const int B = 2048, K = 64;
    float* xfeatn  = (float*)d_ws;            // B*64
    float* scores  = xfeatn + B * 64;         // B*K
    float* weights = scores + B * K;          // B*K
    float* swv     = weights + B * K;         // B

    // 1) normalized x features:  M = 2048 -> 64 super-tiles -> 16 blocks
    feat_gemm_kernel<<<(B / 32) / WPB, TPB, 0, stream>>>(
        x_im, Wtheta, btheta, nullptr, xfeatn, 0);
    // 2) cosine scores: M = 131072 -> 4096 super-tiles -> 1024 blocks (reads p_im once, 403 MB)
    feat_gemm_kernel<<<((B * K) / 32) / WPB, TPB, 0, stream>>>(
        p_im, Wphi, bphi, xfeatn, scores, 1);
    // 3) switch + softmax weights
    softmax_kernel<<<B / 8, 256, 0, stream>>>(scores, sscale, sshift, weights, swv);
    // 4) weighted channel-mix + blend (skips zero-weight k, so p is barely touched)
    output_kernel<<<B, 256, 0, stream>>>(x, p, weights, swv, Wg, bg, Wo, bo, out);
}
